// SNU_Net_76450417869386
// MI455X (gfx1250) — matrix operand order repaired and checked by reading, not compiled, not run on hardware
//
#include <hip/hip_runtime.h>
#include <hip/hip_bf16.h>

typedef __attribute__((ext_vector_type(16))) _Float16 v16h;
typedef __attribute__((ext_vector_type(8)))  float    v8f;

#define T_DATA   10000
#define E_NO     2000
#define I_NO     500
#define SUB_NO   64
#define KERN_LEN 201
#define PAD      100

// ---------------------------------------------------------------------------
// Kernel 1: drive[t,s] = exp(w0[s]) * (S_e @ C_syn_e^T)[t,s]
//                      - exp(w1[s]) * (S_i @ C_syn_i^T)[t,s]
// Exact in f16 WMMA: all A/B elements are 0/1, accumulation is f32.
// Block = 256 threads = 8 waves covering 32 T-rows x all 64 columns.
// ---------------------------------------------------------------------------

// A fragment (16x32 f16), FULL tile: unconditional vector loads.
// k0 already includes half*16. Elements 0..15 <-> K k0..k0+15.
// All addresses 16B-aligned.
__device__ inline v16h fragA_full(const float* __restrict__ p, int k0) {
    const float4* q = (const float4*)(p + k0);
    float4 a0 = q[0], a1 = q[1];      // k0+0..7
    float4 a2 = q[2], a3 = q[3];      // k0+8..15
    v16h r;
    r[0]  = (_Float16)a0.x; r[1]  = (_Float16)a0.y; r[2]  = (_Float16)a0.z; r[3]  = (_Float16)a0.w;
    r[4]  = (_Float16)a1.x; r[5]  = (_Float16)a1.y; r[6]  = (_Float16)a1.z; r[7]  = (_Float16)a1.w;
    r[8]  = (_Float16)a2.x; r[9]  = (_Float16)a2.y; r[10] = (_Float16)a2.z; r[11] = (_Float16)a2.w;
    r[12] = (_Float16)a3.x; r[13] = (_Float16)a3.y; r[14] = (_Float16)a3.z; r[15] = (_Float16)a3.w;
    return r;
}

// A fragment, K-TAIL: clamped-address unconditional load + value select
// (v_cndmask, no exec-mask branching). Runs once per GEMM.
__device__ inline v16h fragA_tail(const float* __restrict__ p, int k0, int Klim) {
    v16h r;
#pragma unroll
    for (int i = 0; i < 16; ++i) {
        int k  = k0 + i;
        int kc = (k < Klim) ? k : (Klim - 1);
        float v = p[kc];
        r[i] = (_Float16)((k < Klim) ? v : 0.0f);
    }
    return r;
}

// B fragment (32x16 f16), FULL tile: 16 consecutive K starting at k0
// (k0 includes half*16). 16B-aligned -> 4x float4.
__device__ inline v16h fragB_full(const float* __restrict__ p, int k0) {
    const float4* q = (const float4*)(p + k0);
    float4 b0 = q[0], b1 = q[1], b2 = q[2], b3 = q[3];
    v16h r;
    r[0]  = (_Float16)b0.x; r[1]  = (_Float16)b0.y; r[2]  = (_Float16)b0.z; r[3]  = (_Float16)b0.w;
    r[4]  = (_Float16)b1.x; r[5]  = (_Float16)b1.y; r[6]  = (_Float16)b1.z; r[7]  = (_Float16)b1.w;
    r[8]  = (_Float16)b2.x; r[9]  = (_Float16)b2.y; r[10] = (_Float16)b2.z; r[11] = (_Float16)b2.w;
    r[12] = (_Float16)b3.x; r[13] = (_Float16)b3.y; r[14] = (_Float16)b3.z; r[15] = (_Float16)b3.w;
    return r;
}

__device__ inline v16h fragB_tail(const float* __restrict__ p, int k0, int Klim) {
    v16h r;
#pragma unroll
    for (int i = 0; i < 16; ++i) {
        int k  = k0 + i;
        int kc = (k < Klim) ? k : (Klim - 1);
        float v = p[kc];
        r[i] = (_Float16)((k < Klim) ? v : 0.0f);
    }
    return r;
}

__global__ __launch_bounds__(256)
void snu_drive_wmma(const float* __restrict__ S_e, const float* __restrict__ S_i,
                    const float* __restrict__ Ce,  const float* __restrict__ Ci,
                    const float* __restrict__ Wraw, float* __restrict__ drive) {
    const int lane  = threadIdx.x & 31;
    const int wave  = threadIdx.x >> 5;
    const int mtile = wave >> 2;        // 0..1
    const int ntile = wave & 3;         // 0..3
    const int half  = lane >> 4;        // 0|1
    const int m0    = blockIdx.x * 32 + mtile * 16;

    const int  mrow  = m0 + (lane & 15);
    const bool mok   = (mrow < T_DATA);
    const int  mrowc = mok ? mrow : (T_DATA - 1);   // clamp: loads stay in-bounds,
                                                    // results discarded by store guard
    const int  n_g   = ntile * 16 + (lane & 15);

    const float* arow_e = S_e + (size_t)mrowc * E_NO;
    const float* arow_i = S_i + (size_t)mrowc * I_NO;
    const float* brow_e = Ce + (size_t)n_g * E_NO;
    const float* brow_i = Ci + (size_t)n_g * I_NO;

    const int ka = half * 16;   // A lane K offset
    const int kb = half * 16;   // B lane K offset

    // ---- S_e GEMM: 62 full K-steps + 1 tail (1984..1999 valid of ..2015)
    v8f acc_e = {};
    constexpr int E_MAIN = (E_NO / 32) * 32;     // 1984
    for (int k0 = 0; k0 < E_MAIN; k0 += 32) {
        v16h a = fragA_full(arow_e, k0 + ka);
        v16h b = fragB_full(brow_e, k0 + kb);
        acc_e = __builtin_amdgcn_wmma_f32_16x16x32_f16(false, a, false, b,
                                                       (short)0, acc_e, false, false);
    }
    {
        v16h a = fragA_tail(arow_e, E_MAIN + ka, E_NO);
        v16h b = fragB_tail(brow_e, E_MAIN + kb, E_NO);
        acc_e = __builtin_amdgcn_wmma_f32_16x16x32_f16(false, a, false, b,
                                                       (short)0, acc_e, false, false);
    }

    // ---- S_i GEMM: 15 full K-steps + 1 tail (480..499 valid of ..511)
    v8f acc_i = {};
    constexpr int I_MAIN = (I_NO / 32) * 32;     // 480
    for (int k0 = 0; k0 < I_MAIN; k0 += 32) {
        v16h a = fragA_full(arow_i, k0 + ka);
        v16h b = fragB_full(brow_i, k0 + kb);
        acc_i = __builtin_amdgcn_wmma_f32_16x16x32_f16(false, a, false, b,
                                                       (short)0, acc_i, false, false);
    }
    {
        v16h a = fragA_tail(arow_i, I_MAIN + ka, I_NO);
        v16h b = fragB_tail(brow_i, I_MAIN + kb, I_NO);
        acc_i = __builtin_amdgcn_wmma_f32_16x16x32_f16(false, a, false, b,
                                                       (short)0, acc_i, false, false);
    }

    const float we =  __expf(Wraw[2 * n_g + 0]);
    const float wi = -__expf(Wraw[2 * n_g + 1]);

#pragma unroll
    for (int r = 0; r < 8; ++r) {
        int m = m0 + half * 8 + r;               // C/D layout: M = r + 8*half
        if (m < T_DATA)
            drive[(size_t)m * SUB_NO + n_g] = acc_e[r] * we + acc_i[r] * wi;
    }
}

// ---------------------------------------------------------------------------
// Kernel 2: serial scan, one wave32, lane l owns subunits 2l and 2l+1.
// Coupling (spk*prop) @ C_den^T handled sparsely: spikes -> wave-uniform
// ballot masks -> uniform loop over set bits, ds_load_b64 of precomputed
// P[j][s] = prop[j]*C_den[s][j] (stored [j][s] so lanes read contiguously).
// ---------------------------------------------------------------------------
__global__ __launch_bounds__(32)
void snu_scan(const float* __restrict__ drive, const float* __restrict__ C_den,
              const float* __restrict__ dec_raw, const float* __restrict__ thr_raw,
              const float* __restrict__ prop_raw, float* __restrict__ spk_out) {
    __shared__ __align__(16) float Pt[SUB_NO * SUB_NO];   // [j][s], 16 KB
    const int l = threadIdx.x;

    for (int idx = l; idx < SUB_NO * SUB_NO; idx += 32) {
        int j = idx >> 6, s = idx & 63;
        Pt[idx] = __expf(prop_raw[j]) * C_den[s * SUB_NO + j];
    }
    __syncthreads();

    const int s0 = 2 * l, s1 = 2 * l + 1;
    const float dec0 = 1.0f / (1.0f + __expf(-dec_raw[s0]));
    const float dec1 = 1.0f / (1.0f + __expf(-dec_raw[s1]));
    const float th0  = -__expf(thr_raw[s0]);
    const float th1  = -__expf(thr_raw[s1]);

    const float2* drv2 = (const float2*)drive;
    const float2* Pt2  = (const float2*)Pt;
    float2*       out2 = (float2*)spk_out;

    float V0 = 0.f, V1 = 0.f, spk0 = 0.f, spk1 = 0.f;
    float2 d = drv2[l];                                   // prefetch t=0

    for (int t = 0; t < T_DATA; ++t) {
        float2 dn = d;
        if (t + 1 < T_DATA) dn = drv2[(size_t)(t + 1) * 32 + l];  // hide latency

        unsigned me = (unsigned)__ballot(spk0 > 0.5f);    // bit l -> spk[2l]
        unsigned mo = (unsigned)__ballot(spk1 > 0.5f);    // bit l -> spk[2l+1]

        float c0 = 0.f, c1 = 0.f;
        while (me) {                                      // uniform loop
            int b = __ffs(me) - 1; me &= me - 1;
            float2 p = Pt2[(size_t)(2 * b) * 32 + l];
            c0 += p.x; c1 += p.y;
        }
        while (mo) {
            int b = __ffs(mo) - 1; mo &= mo - 1;
            float2 p = Pt2[(size_t)(2 * b + 1) * 32 + l];
            c0 += p.x; c1 += p.y;
        }

        V0 = V0 * (1.0f - spk0) * dec0 + d.x + c0;
        V1 = V1 * (1.0f - spk1) * dec1 + d.y + c1;
        spk0 = (V0 + th0 > 0.0f) ? 1.0f : 0.0f;           // heaviside(x,0)
        spk1 = (V1 + th1 > 0.0f) ? 1.0f : 0.0f;

        out2[(size_t)t * 32 + l] = make_float2(spk0, spk1);
        d = dn;
    }
}

// ---------------------------------------------------------------------------
// Kernel 3: final[i] = sum_k spk_out[i+k-PAD, 0] * kern[KERN_LEN-1-k]
// kern[m] = (m*e^-2) * exp(-m*e^-2) * e^0.75
// ---------------------------------------------------------------------------
__global__ __launch_bounds__(256)
void snu_conv(const float* __restrict__ spk_out, float* __restrict__ final_out) {
    __shared__ float wflip[KERN_LEN];
    const int tid = threadIdx.x;
    if (tid < KERN_LEN) {
        float m  = (float)(KERN_LEN - 1 - tid);
        float tt = m * 0.13533528323661270f;              // exp(-2)
        wflip[tid] = tt * __expf(-tt) * 2.11700001661267470f; // exp(0.75)
    }
    __syncthreads();

    int i = blockIdx.x * blockDim.x + tid;
    if (i >= T_DATA) return;
    float s = 0.f;
#pragma unroll 4
    for (int k = 0; k < KERN_LEN; ++k) {
        int idx = i + k - PAD;
        if (idx >= 0 && idx < T_DATA)
            s += spk_out[(size_t)idx * SUB_NO] * wflip[k];
    }
    final_out[i] = s;
}

// ---------------------------------------------------------------------------
extern "C" void kernel_launch(void* const* d_in, const int* in_sizes, int n_in,
                              void* d_out, int out_size, void* d_ws, size_t ws_size,
                              hipStream_t stream) {
    (void)in_sizes; (void)n_in; (void)out_size; (void)ws_size;
    const float* S_e   = (const float*)d_in[0];
    const float* S_i   = (const float*)d_in[1];
    const float* C_den = (const float*)d_in[2];
    const float* Ce    = (const float*)d_in[3];
    const float* Ci    = (const float*)d_in[4];
    const float* Wraw  = (const float*)d_in[5];
    const float* decR  = (const float*)d_in[6];
    const float* thrR  = (const float*)d_in[7];
    const float* prpR  = (const float*)d_in[8];

    float* final_out = (float*)d_out;            // [0 .. T)
    float* spk_out   = (float*)d_out + T_DATA;   // [T .. T + T*64)
    float* drive     = (float*)d_ws;             // T*64 f32 scratch

    dim3 grid_g((T_DATA + 31) / 32);             // 313 blocks, 8 waves each
    snu_drive_wmma<<<grid_g, 256, 0, stream>>>(S_e, S_i, Ce, Ci, Wraw, drive);

    snu_scan<<<1, 32, 0, stream>>>(drive, C_den, decR, thrR, prpR, spk_out);

    snu_conv<<<(T_DATA + 255) / 256, 256, 0, stream>>>(spk_out, final_out);
}
